// BasicBlock_16363825397983
// MI455X (gfx1250) — matrix operand order repaired and checked by reading, not compiled, not run on hardware
//
#include <hip/hip_runtime.h>

// ---------------------------------------------------------------------------
// ISTA-Net+ CT block for MI455X (gfx1250, wave32).
// Heavy 32x32ch 3x3 convs -> implicit GEMM via v_wmma_f32_16x16x32_bf16 on
// halo-padded bf16 NHWC activations (no predication, immediate-offset loads).
// Radon fwd / ramp filter: one wave per output, lane-strided + shfl reduction.
// Backproj + 1<->32ch convs + elementwise -> f32 VALU.
// ---------------------------------------------------------------------------

#define IMG   512
#define PS    (IMG + 2)          // padded span (1-pixel zero halo)
#define NANG  90
#define DET   729
#define FEAT  32
#define PI_F  3.14159265358979323846f

typedef __attribute__((ext_vector_type(16))) __bf16 v16bf;
typedef __attribute__((ext_vector_type(8)))  float  v8f;
typedef __attribute__((ext_vector_type(4)))  int    vi4;

union Frag {
    v16bf v;
    vi4   q[2];
};

// float -> bf16 bits, round-to-nearest-even
__device__ inline unsigned short f2bf(float f) {
    unsigned int u = __float_as_uint(f);
    unsigned int lsb = (u >> 16) & 1u;
    u += 0x7fffu + lsb;
    return (unsigned short)(u >> 16);
}
__device__ inline float bf2f(unsigned short h) {
    return __uint_as_float(((unsigned int)h) << 16);
}

// 5-step butterfly reduction across a wave32
__device__ inline float wave_sum(float v) {
#pragma unroll
    for (int o = 16; o > 0; o >>= 1) v += __shfl_xor(v, o, 32);
    return v;
}

// ---------------------------------------------------------------------------
// 0a) zero the 1-pixel halo border of a padded NHWC bf16 buffer
// ---------------------------------------------------------------------------
__global__ void zero_border_k(unsigned short* __restrict__ b) {
    int i = blockIdx.x * blockDim.x + threadIdx.x;
    if (i >= PS * PS) return;
    int y = i / PS, x = i % PS;
    if (y == 0 || y == PS - 1 || x == 0 || x == PS - 1) {
        vi4 z = {0, 0, 0, 0};
        vi4* p = reinterpret_cast<vi4*>(b + (size_t)i * FEAT);
        p[0] = z; p[1] = z; p[2] = z; p[3] = z;
    }
}

// 0b) precompute cos/sin tables for the 90 angles
__global__ void trig_k(const float* __restrict__ theta,
                       float* __restrict__ cs, float* __restrict__ sn) {
    int a = blockIdx.x * blockDim.x + threadIdx.x;
    if (a < NANG) { cs[a] = cosf(theta[a]); sn[a] = sinf(theta[a]); }
}

// ---------------------------------------------------------------------------
// 1) Radon forward: ONE WAVE per (angle, detector) ray; lanes stride the
//    729 bilinear taps by 32, then wave-reduce. 65610 waves fill the machine.
// ---------------------------------------------------------------------------
__global__ void radon_fwd_k(const float* __restrict__ img,
                            const float* __restrict__ cst,
                            const float* __restrict__ snt,
                            float* __restrict__ sino) {
    int gw = blockIdx.x * 8 + (threadIdx.x >> 5);   // global wave id
    int ln = threadIdx.x & 31;
    if (gw >= NANG * DET) return;
    int a = gw / DET, i = gw % DET;
    float cs = cst[a], sn = snt[a];
    float si = (float)i - 364.0f;   // (DET-1)/2
    float bx = 255.5f + si * cs;
    float by = 255.5f + si * sn;
    float acc = 0.0f;
    for (int j = ln; j < DET; j += 32) {
        float sj = (float)j - 364.0f;
        float px = bx - sj * sn;
        float py = by + sj * cs;
        float fx0 = floorf(px), fy0 = floorf(py);
        int x0 = (int)fx0, y0 = (int)fy0;
        float fx = px - fx0, fy = py - fy0;
        bool xa = (x0 >= 0) && (x0 < IMG);
        bool xb = (x0 + 1 >= 0) && (x0 + 1 < IMG);
        bool ya = (y0 >= 0) && (y0 < IMG);
        bool yb = (y0 + 1 >= 0) && (y0 + 1 < IMG);
        float v00 = (xa && ya) ? img[y0 * IMG + x0] : 0.0f;
        float v01 = (xb && ya) ? img[y0 * IMG + x0 + 1] : 0.0f;
        float v10 = (xa && yb) ? img[(y0 + 1) * IMG + x0] : 0.0f;
        float v11 = (xb && yb) ? img[(y0 + 1) * IMG + x0 + 1] : 0.0f;
        acc += (1.0f - fy) * ((1.0f - fx) * v00 + fx * v01)
             +          fy * ((1.0f - fx) * v10 + fx * v11);
    }
    acc = wave_sum(acc);
    if (ln == 0) sino[gw] = acc;
}

// ---------------------------------------------------------------------------
// 2) Ramp filter: direct linear (non-wrapping) convolution, closed-form kernel
//    g[0]=0.5, g[odd m]=-2/(pi*m)^2, 0 otherwise. ONE WAVE per output sample,
//    lane-strided over k, wave-reduce; then scale and subtract measurement.
// ---------------------------------------------------------------------------
__global__ void filt_k(const float* __restrict__ sino,
                       const float* __restrict__ sino_meas,
                       float* __restrict__ filt) {
    int gw = blockIdx.x * 8 + (threadIdx.x >> 5);
    int ln = threadIdx.x & 31;
    if (gw >= NANG * DET) return;
    int a = gw / DET, d = gw % DET;
    const float* row = sino + a * DET;
    float acc = 0.0f;
    for (int k = ln; k < DET; k += 32) {
        int m = d - k; m = (m < 0) ? -m : m;
        float w;
        if (m == 0)      w = 0.5f;
        else if (m & 1)  w = -2.0f / (PI_F * PI_F * (float)(m * m));
        else             w = 0.0f;
        acc += row[k] * w;
    }
    acc = wave_sum(acc);
    if (ln == 0) filt[gw] = acc * (PI_F / (2.0f * (float)NANG)) - sino_meas[gw];
}

// ---------------------------------------------------------------------------
// 3) Backprojection + gradient step: x_input = x - lambda * BP(filt)
// ---------------------------------------------------------------------------
__global__ void backproj_k(const float* __restrict__ filt,
                           const float* __restrict__ cst,
                           const float* __restrict__ snt,
                           const float* __restrict__ x,
                           const float* __restrict__ lam,
                           float* __restrict__ x_in) {
    int i = blockIdx.x * blockDim.x + threadIdx.x;
    if (i >= IMG * IMG) return;
    int yy = i >> 9, xx = i & 511;
    float xs = (float)xx - 255.5f, ys = (float)yy - 255.5f;
    float acc = 0.0f;
    for (int a = 0; a < NANG; ++a) {
        float pos = xs * cst[a] + ys * snt[a] + 364.0f;
        int i0 = (int)floorf(pos);
        i0 = (i0 < 0) ? 0 : ((i0 > DET - 2) ? DET - 2 : i0);
        float w = pos - (float)i0;
        const float* row = filt + a * DET;
        float v = row[i0] * (1.0f - w) + row[i0 + 1] * w;
        acc += ((pos >= 0.0f) && (pos <= (float)(DET - 1))) ? v : 0.0f;
    }
    x_in[i] = x[i] - lam[0] * acc;
}

// ---------------------------------------------------------------------------
// 4) conv_D (1 -> 32): emit f32 NCHW (for symloss) + padded bf16 NHWC
// ---------------------------------------------------------------------------
__global__ void convD_k(const float* __restrict__ x_in,
                        const float* __restrict__ wD,
                        float* __restrict__ xD_f32,
                        unsigned short* __restrict__ xD_bf) {
    int i = blockIdx.x * blockDim.x + threadIdx.x;
    if (i >= IMG * IMG) return;
    int y = i >> 9, x = i & 511;
    float v[9];
#pragma unroll
    for (int s = 0; s < 9; ++s) {
        int yy = y + s / 3 - 1, xx = x + s % 3 - 1;
        v[s] = (yy >= 0 && yy < IMG && xx >= 0 && xx < IMG) ? x_in[yy * IMG + xx] : 0.0f;
    }
    unsigned short* ob = xD_bf + ((size_t)(y + 1) * PS + (x + 1)) * FEAT;
    for (int oc = 0; oc < FEAT; ++oc) {
        float acc = 0.0f;
#pragma unroll
        for (int s = 0; s < 9; ++s) acc += v[s] * wD[oc * 9 + s];
        xD_f32[oc * (IMG * IMG) + i] = acc;
        ob[oc] = f2bf(acc);
    }
}

// ---------------------------------------------------------------------------
// 5) 32->32 3x3 conv, implicit GEMM on v_wmma_f32_16x16x32_bf16.
//    Wave owns 32 pixels x 32 out-channels = four 16x16 tiles, 9 K-slices
//    (36 WMMAs). All A taps are constant byte offsets from one per-lane base
//    (zero halo) -> global_load_b128 with immediate offsets, no address VALU.
//    MODE 0: padded bf16 NHWC out (ACT: 0 none, 1 relu)
//    MODE 3: f32 (acc - sub[]) duplicated into oA and oB (symloss path)
// ---------------------------------------------------------------------------
template <int MODE, int ACT>
__global__ void conv32_k(const unsigned short* __restrict__ in,  // padded NHWC
                         const float* __restrict__ w,
                         unsigned short* __restrict__ obf,       // padded NHWC
                         float* __restrict__ oA,
                         float* __restrict__ oB,
                         const float* __restrict__ sub) {
    __shared__ unsigned short wlds[FEAT * 9 * FEAT];   // [oc][tap][ic], 18 KB
    int tid = threadIdx.x;
    for (int idx = tid; idx < FEAT * 9 * FEAT; idx += 256) {
        int oc = idx / (9 * FEAT);
        int r  = idx % (9 * FEAT);
        int s  = r / FEAT;
        int ic = r % FEAT;
        wlds[idx] = f2bf(w[(oc * FEAT + ic) * 9 + s]);
    }
    __syncthreads();

    int wv = tid >> 5, ln = tid & 31;
    int gw = blockIdx.x * 8 + wv;          // 8192 waves total
    int y  = gw >> 4;                      // 16 x-tiles (of 32 px) per row
    int xb = (gw & 15) << 5;
    int half = ln >> 4;                    // K-half per 16-bit A/B layout
    int lp   = ln & 15;

    // single per-lane bases; all taps become compile-time immediate offsets
    const unsigned short* abase = in + ((size_t)(y * PS + xb + lp) * FEAT + half * 8);
    const unsigned short* wbase = &wlds[(lp * 9) * FEAT + half * 8];

    v8f acc00 = {}, acc01 = {}, acc10 = {}, acc11 = {};

#pragma unroll
    for (int ky = 0; ky < 3; ++ky) {
#pragma unroll
        for (int kx = 0; kx < 3; ++kx) {
            const int s = ky * 3 + kx;
            const int aoff = (ky * PS + kx) * FEAT;        // constant per tap

            Frag a0, a1;
            // A 16x32 bf16 layout: ch bytes [half*16,+16) and [half*16+32,+16)
            const vi4* p0 = reinterpret_cast<const vi4*>(abase + aoff);
            a0.q[0] = p0[0];
            a0.q[1] = p0[2];
            const vi4* p1 = reinterpret_cast<const vi4*>(abase + aoff + 16 * FEAT);
            a1.q[0] = p1[0];
            a1.q[1] = p1[2];

            // B 32x16 bf16 layout: input channels [half*8,+8) and [16+half*8,+8) for one oc
            Frag b0, b1;
            const vi4* bp0 = reinterpret_cast<const vi4*>(wbase + s * FEAT);
            b0.q[0] = bp0[0]; b0.q[1] = bp0[2];
            const vi4* bp1 = reinterpret_cast<const vi4*>(wbase + (16 * 9 + s) * FEAT);
            b1.q[0] = bp1[0]; b1.q[1] = bp1[2];

            acc00 = __builtin_amdgcn_wmma_f32_16x16x32_bf16(false, a0.v, false, b0.v,
                                                            (short)0, acc00, false, false);
            acc01 = __builtin_amdgcn_wmma_f32_16x16x32_bf16(false, a0.v, false, b1.v,
                                                            (short)0, acc01, false, false);
            acc10 = __builtin_amdgcn_wmma_f32_16x16x32_bf16(false, a1.v, false, b0.v,
                                                            (short)0, acc10, false, false);
            acc11 = __builtin_amdgcn_wmma_f32_16x16x32_bf16(false, a1.v, false, b1.v,
                                                            (short)0, acc11, false, false);
        }
    }

#pragma unroll
    for (int m = 0; m < 2; ++m) {
#pragma unroll
        for (int r = 0; r < 8; ++r) {
            int xx = xb + m * 16 + r + half * 8;   // C/D layout: M = r + 8*(lane>=16)
            float v0 = (m == 0) ? acc00[r] : acc10[r];
            float v1 = (m == 0) ? acc01[r] : acc11[r];
            if (ACT == 1) { v0 = fmaxf(v0, 0.0f); v1 = fmaxf(v1, 0.0f); }
            int oc0 = lp, oc1 = lp + 16;           // C/D layout: N = lane & 15
            if (MODE == 0) {
                unsigned short* ob = obf + ((size_t)(y + 1) * PS + (xx + 1)) * FEAT;
                ob[oc0] = f2bf(v0);
                ob[oc1] = f2bf(v1);
            } else {  // MODE == 3: symloss written twice
                int pix = y * IMG + xx;
                int i0 = oc0 * (IMG * IMG) + pix;
                int i1 = oc1 * (IMG * IMG) + pix;
                float s0 = v0 - sub[i0];
                float s1 = v1 - sub[i1];
                oA[i0] = s0; oB[i0] = s0;
                oA[i1] = s1; oB[i1] = s1;
            }
        }
    }
}

// ---------------------------------------------------------------------------
// 6) soft threshold over the full padded buffer (softthr(0) == 0 keeps halo)
// ---------------------------------------------------------------------------
__global__ void softthr_k(const unsigned short* __restrict__ in,
                          const float* __restrict__ thr,
                          unsigned short* __restrict__ out, int n) {
    int i = blockIdx.x * blockDim.x + threadIdx.x;
    if (i >= n) return;
    float v = bf2f(in[i]);
    float m = fabsf(v) - thr[0];
    m = (m > 0.0f) ? m : 0.0f;
    float r = (v > 0.0f) ? m : ((v < 0.0f) ? -m : 0.0f);
    out[i] = f2bf(r);
}

// ---------------------------------------------------------------------------
// 7) conv_G (32 -> 1) + residual: x_pred = x_input + conv_G(x_backward)
//    Reads the padded buffer -> no bounds checks needed.
// ---------------------------------------------------------------------------
__global__ void convG_k(const unsigned short* __restrict__ xbwd,  // padded NHWC
                        const float* __restrict__ wG,
                        const float* __restrict__ x_in,
                        float* __restrict__ out) {
    int i = blockIdx.x * blockDim.x + threadIdx.x;
    if (i >= IMG * IMG) return;
    int y = i >> 9, x = i & 511;
    float acc = 0.0f;
#pragma unroll
    for (int s = 0; s < 9; ++s) {
        const unsigned short* p = xbwd + ((size_t)(y + s / 3) * PS + (x + s % 3)) * FEAT;
        for (int ic = 0; ic < FEAT; ++ic) acc += bf2f(p[ic]) * wG[ic * 9 + s];
    }
    out[i] = x_in[i] + acc;
}

// ---------------------------------------------------------------------------
extern "C" void kernel_launch(void* const* d_in, const int* in_sizes, int n_in,
                              void* d_out, int out_size, void* d_ws, size_t ws_size,
                              hipStream_t stream) {
    (void)in_sizes; (void)n_in; (void)out_size; (void)ws_size;
    const float* x         = (const float*)d_in[0];
    const float* theta     = (const float*)d_in[1];
    const float* sino_meas = (const float*)d_in[2];
    const float* lam       = (const float*)d_in[3];
    const float* thr       = (const float*)d_in[4];
    const float* wD        = (const float*)d_in[5];
    const float* w1f       = (const float*)d_in[6];
    const float* w2f       = (const float*)d_in[7];
    const float* w1b       = (const float*)d_in[8];
    const float* w2b       = (const float*)d_in[9];
    const float* wG        = (const float*)d_in[10];
    float* out = (float*)d_out;

    char* ws = (char*)d_ws;
    size_t off = 0;
    auto alloc = [&](size_t bytes) -> char* {
        char* p = ws + off;
        off = (off + bytes + 255) & ~(size_t)255;
        return p;
    };
    float* cst              = (float*)alloc((size_t)NANG * 4);
    float* snt              = (float*)alloc((size_t)NANG * 4);
    float* s_sino           = (float*)alloc((size_t)NANG * DET * 4);
    float* s_filt           = (float*)alloc((size_t)NANG * DET * 4);
    float* x_in             = (float*)alloc((size_t)IMG * IMG * 4);
    float* xD_f32           = (float*)alloc((size_t)FEAT * IMG * IMG * 4);
    unsigned short* b0      = (unsigned short*)alloc((size_t)FEAT * PS * PS * 2);
    unsigned short* b1      = (unsigned short*)alloc((size_t)FEAT * PS * PS * 2);
    unsigned short* b2      = (unsigned short*)alloc((size_t)FEAT * PS * PS * 2);

    const int T = 256;
    const int nRayW = (NANG * DET + 7) / 8;          // wave-per-output kernels
    const int nPix  = (IMG * IMG + T - 1) / T;
    const int nPad  = (PS * PS + T - 1) / T;
    const int nPadE = (FEAT * PS * PS + T - 1) / T;
    const int nConv = (IMG * (IMG / 32)) / 8;        // 8192 waves / 8 per block

    // Halo + trig prep
    zero_border_k<<<nPad, T, 0, stream>>>(b0);
    zero_border_k<<<nPad, T, 0, stream>>>(b1);
    zero_border_k<<<nPad, T, 0, stream>>>(b2);
    trig_k<<<1, 128, 0, stream>>>(theta, cst, snt);

    // Physics pipeline (f32 VALU, wave-parallel reductions)
    radon_fwd_k<<<nRayW, T, 0, stream>>>(x, cst, snt, s_sino);
    filt_k<<<nRayW, T, 0, stream>>>(s_sino, sino_meas, s_filt);
    backproj_k<<<nPix, T, 0, stream>>>(s_filt, cst, snt, x, lam, x_in);

    // Learned prior (bf16 WMMA)
    convD_k<<<nPix, T, 0, stream>>>(x_in, wD, xD_f32, b0);
    conv32_k<0, 1><<<nConv, T, 0, stream>>>(b0, w1f, b1, nullptr, nullptr, nullptr); // relu(conv1_f(x_D))
    conv32_k<0, 0><<<nConv, T, 0, stream>>>(b1, w2f, b2, nullptr, nullptr, nullptr); // x_forward
    softthr_k<<<nPadE, T, 0, stream>>>(b2, thr, b0, FEAT * PS * PS);                 // st (keeps halo zero)
    conv32_k<0, 1><<<nConv, T, 0, stream>>>(b0, w1b, b1, nullptr, nullptr, nullptr); // relu(conv1_b(st))
    conv32_k<0, 0><<<nConv, T, 0, stream>>>(b1, w2b, b0, nullptr, nullptr, nullptr); // x_backward
    convG_k<<<nPix, T, 0, stream>>>(b0, wG, x_in, out);                              // x_pred

    // Symmetry loss branch: x_D_est - x_D written to both output slots
    conv32_k<0, 1><<<nConv, T, 0, stream>>>(b2, w1b, b1, nullptr, nullptr, nullptr); // relu(conv1_b(x_forward))
    conv32_k<3, 0><<<nConv, T, 0, stream>>>(b1, w2b, nullptr,
                                            out + (size_t)IMG * IMG,
                                            out + (size_t)IMG * IMG + (size_t)FEAT * IMG * IMG,
                                            xD_f32);
}
